// Competitive_Progressive_Temporal_Module_28664611734058
// MI455X (gfx1250) — matrix operand order repaired and checked by reading, not compiled, not run on hardware
//
#include <hip/hip_runtime.h>
#include <hip/hip_bf16.h>
#include <stdint.h>

// ---------------------------------------------------------------------------
// Competitive Progressive Temporal Module for MI455X (gfx1250, wave32, WMMA)
//   x:(8,64,16,56,56) f32 -> 3 progressive temporal convs (3x1x1, SAME,+ReLU)
//   -> GAP -> FC(64->32)+BN+ReLU -> per-branch attention (softmax over 3)
//   -> weighted branch sum -> (128,64,56,56) f32
// ---------------------------------------------------------------------------

typedef __attribute__((ext_vector_type(16))) _Float16 v16h;
typedef __attribute__((ext_vector_type(8)))  _Float16 h8;
typedef __attribute__((ext_vector_type(4)))  _Float16 h4;
typedef __attribute__((ext_vector_type(8)))  float    v8f;

#define NB    8          // batch
#define CH    64         // channels
#define TT    16         // temporal length
#define HWD   3136       // 56*56
#define TILE  16         // spatial pixels per block (WMMA N)
#define NTILE (HWD / TILE)   // 196
#define DD    32         // bottleneck dim
#define NBR   3          // branches

// dynamic LDS carve-up (halfs):
//   buf0 / buf1 : [18][16][64] f16  (t halo rows 0 and 17 stay zero = SAME pad)
//   wlds        : [3][3][64][64] f16 = [stage][tap][co][ci]
//   bias        : [3][64] f32
#define BUF_H   (18 * 16 * 64)
#define WLDS_H  (3 * 3 * 64 * 64)
#define SMEM_BYTES (2 * BUF_H * 2 + WLDS_H * 2 + 3 * 64 * 4)

// ---------------------------------------------------------------------------
// Phase 1: fused progressive WMMA convolution
//   grid = NB*NTILE blocks, 128 threads (4 waves; wave w owns out-chans 16w..)
// ---------------------------------------------------------------------------
__global__ __launch_bounds__(128)
void cptm_conv_kernel(const float* __restrict__ x,
                      const float* __restrict__ conv_w,   // (3,64,64,3,1,1)
                      const float* __restrict__ conv_b,   // (3,64)
                      _Float16* __restrict__ branch,      // [3][8][64][16][3136] f16
                      float* __restrict__ partial)        // [NTILE][8][64] f32
{
    extern __shared__ _Float16 smem[];
    _Float16* buf0 = smem;
    _Float16* buf1 = smem + BUF_H;
    _Float16* wlds = smem + 2 * BUF_H;
    float*    bias = (float*)(smem + 2 * BUF_H + WLDS_H);

    const int tid  = threadIdx.x;
    const int lane = tid & 31;
    const int wave = tid >> 5;
    const int n    = blockIdx.x / NTILE;
    const int tile = blockIdx.x % NTILE;
    const int hw0  = tile * TILE;

    // ---- stage weights into LDS as [st][tap][co][ci] f16 (ci contiguous) ----
    for (int j = tid; j < WLDS_H; j += 128) {
        int ci  = j & 63;
        int co  = (j >> 6) & 63;
        int st3 = j >> 12;              // st*3 + tap
        int tap = st3 % 3, st = st3 / 3;
        wlds[j] = (_Float16)conv_w[((size_t)(st * 64 + co) * 64 + ci) * 3 + tap];
    }
    for (int j = tid; j < 3 * 64; j += 128) bias[j] = conv_b[j];

    // ---- zero temporal halo rows tt=0 and tt=17 in both ping-pong buffers ----
    for (int j = tid; j < 2 * 16 * 64; j += 128) {
        int r  = j & 1023;              // (s,c)
        int tt = (j < 1024) ? 0 : 17;
        buf0[tt * 16 * 64 + r] = (_Float16)0.f;
        buf1[tt * 16 * 64 + r] = (_Float16)0.f;
    }

    // ---- stage x tile: buf0[t+1][s][c] = (f16) x[n][c][t][hw0+s] ----
    for (int p = tid; p < CH * TT; p += 128) {
        int c = p & 63, t = p >> 6;
        const float4* src4 = (const float4*)(x + (((size_t)n * CH + c) * TT + t) * HWD + hw0);
        _Float16* dst = buf0 + ((t + 1) * 16) * 64 + c;
        #pragma unroll
        for (int q = 0; q < 4; ++q) {
            float4 f = src4[q];
            dst[(q * 4 + 0) * 64] = (_Float16)f.x;
            dst[(q * 4 + 1) * 64] = (_Float16)f.y;
            dst[(q * 4 + 2) * 64] = (_Float16)f.z;
            dst[(q * 4 + 3) * 64] = (_Float16)f.w;
        }
    }
    __syncthreads();

    const int m       = lane & 15;      // A-row / B-col / D-col coordinate
    const int halfsel = lane >> 4;
    const int co_base = wave * 16;

    float gacc[8];
    #pragma unroll
    for (int r = 0; r < 8; ++r) gacc[r] = 0.f;

    _Float16* cur = buf0;
    _Float16* nxt = buf1;

    for (int st = 0; st < NBR; ++st) {
        // ---- preload 6 A (weight) fragments for this wave's 16 out-channels.
        // ISA 16-bit A 16x32 layout: a[j] <-> K = (j&7) + (j>>3)*16 + half*8
        v16h afr[6];
        #pragma unroll
        for (int kc = 0; kc < 6; ++kc) {
            int tap = kc >> 1, coff = (kc & 1) << 5;
            const _Float16* wb =
                wlds + ((size_t)((st * 3 + tap) * 64 + co_base + m)) * 64 + coff + halfsel * 8;
            h8 lo = *(const h8*)wb;           // K = half*8 + 0..7
            h8 hi = *(const h8*)(wb + 16);    // K = 16 + half*8 + 0..7
            afr[kc] = __builtin_shufflevector(lo, hi,
                       0,1,2,3,4,5,6,7,8,9,10,11,12,13,14,15);
        }

        for (int t = 0; t < TT; ++t) {
            v8f acc = {};
            // ---- K = 192 = 6 chunks of 32 (tap-major, then 32 in-channels) ----
            #pragma unroll
            for (int kc = 0; kc < 6; ++kc) {
                int tap = kc >> 1, coff = (kc & 1) << 5;
                // B 32x16 layout: b[j] <-> K = (j&7) + (j>>3)*16 + half*8 (N = lane&15)
                const _Float16* bb =
                    cur + ((t + tap) * 16 + m) * 64 + coff + halfsel * 8;
                h8 lo = *(const h8*)bb;
                h8 hi = *(const h8*)(bb + 16);
                v16h bfr = __builtin_shufflevector(lo, hi,
                            0,1,2,3,4,5,6,7,8,9,10,11,12,13,14,15);
                acc = __builtin_amdgcn_wmma_f32_16x16x32_f16(
                          /*neg_a=*/false, afr[kc], /*neg_b=*/false, bfr,
                          /*c_mod=*/(short)0, acc,
                          /*reuse_a=*/false, /*reuse_b=*/false);
            }

            // ---- epilogue: bias + ReLU, GAP accumulate, f16 spill ----
            // D layout: lane holds (M = r + half*8, N = m) for r=0..7
            h8 pack;
            #pragma unroll
            for (int r = 0; r < 8; ++r) {
                int co = co_base + halfsel * 8 + r;
                float v = acc[r] + bias[st * 64 + co];
                v = v > 0.f ? v : 0.f;
                gacc[r] += v;
                pack[r] = (_Float16)v;
                branch[((((size_t)st * NB + n) * CH + co) * TT + t) * HWD + hw0 + m] = pack[r];
            }
            // next-stage input (rows 1..16 only -> halos stay zero)
            *(h8*)(nxt + ((t + 1) * 16 + m) * 64 + co_base + halfsel * 8) = pack;
        }
        __syncthreads();                // all waves finished writing nxt
        _Float16* tmp = cur; cur = nxt; nxt = tmp;
    }

    // ---- deterministic GAP partials: reduce over the 16 N-lanes per half ----
    #pragma unroll
    for (int r = 0; r < 8; ++r) {
        float v = gacc[r];
        v += __shfl_xor(v, 1, 32);
        v += __shfl_xor(v, 2, 32);
        v += __shfl_xor(v, 4, 32);
        v += __shfl_xor(v, 8, 32);
        if (m == 0) {
            int co = co_base + halfsel * 8 + r;
            partial[((size_t)tile * NB + n) * CH + co] = v;
        }
    }
}

// ---------------------------------------------------------------------------
// Phase 2: GAP finish -> FC -> BN -> ReLU -> attention logits -> softmax
// ---------------------------------------------------------------------------
__global__ __launch_bounds__(256)
void cptm_attn_kernel(const float* __restrict__ partial,   // [NTILE][8][64]
                      const float* __restrict__ fc_w,      // (64,32)
                      const float* __restrict__ bn_g, const float* __restrict__ bn_b,
                      const float* __restrict__ bn_m, const float* __restrict__ bn_v,
                      const float* __restrict__ fcs_w,     // (3,32,64)
                      const float* __restrict__ fcs_b,     // (3,64)
                      float* __restrict__ att)             // [8][3][64]
{
    __shared__ float s_fea[NB * CH];        // 512
    __shared__ float s_h[NB * DD];          // 256
    __shared__ float s_lg[NB * NBR * CH];   // 1536
    const int tid = threadIdx.x;
    const float inv = 1.f / (3.f * TT * HWD);   // mean of branch-sum / 3

    for (int i = tid; i < NB * CH; i += 256) {
        int nn = i >> 6, c = i & 63;
        float acc = 0.f;
        for (int tl = 0; tl < NTILE; ++tl)
            acc += partial[((size_t)tl * NB + nn) * CH + c];
        s_fea[i] = acc * inv;
    }
    __syncthreads();

    for (int i = tid; i < NB * DD; i += 256) {
        int nn = i >> 5, d = i & 31;
        float acc = 0.f;
        for (int c = 0; c < CH; ++c) acc += s_fea[nn * CH + c] * fc_w[c * DD + d];
        acc = (acc - bn_m[d]) * rsqrtf(bn_v[d] + 1e-3f) * bn_g[d] + bn_b[d];
        s_h[i] = acc > 0.f ? acc : 0.f;
    }
    __syncthreads();

    for (int i = tid; i < NB * NBR * CH; i += 256) {
        int c  = i & 63;
        int b  = (i >> 6) % 3;
        int nn = i / (NBR * CH);
        float acc = fcs_b[b * CH + c];
        for (int d = 0; d < DD; ++d) acc += s_h[nn * DD + d] * fcs_w[(b * DD + d) * CH + c];
        s_lg[i] = acc;                       // GAMMA == 1.0
    }
    __syncthreads();

    for (int i = tid; i < NB * CH; i += 256) {
        int nn = i >> 6, c = i & 63;
        float l0 = s_lg[(nn * 3 + 0) * CH + c];
        float l1 = s_lg[(nn * 3 + 1) * CH + c];
        float l2 = s_lg[(nn * 3 + 2) * CH + c];
        float mx = fmaxf(l0, fmaxf(l1, l2));
        float e0 = __expf(l0 - mx), e1 = __expf(l1 - mx), e2 = __expf(l2 - mx);
        float is = 1.f / (e0 + e1 + e2);
        att[(nn * 3 + 0) * CH + c] = e0 * is;
        att[(nn * 3 + 1) * CH + c] = e1 * is;
        att[(nn * 3 + 2) * CH + c] = e2 * is;
    }
}

// ---------------------------------------------------------------------------
// Phase 3: streaming weighted branch mix.  out[(n*T+t),c,hw] = sum_b a_b * f_b
//   q decomposition order (hw4,c,t,n) == output layout -> coalesced store at q
// ---------------------------------------------------------------------------
__global__ __launch_bounds__(256)
void cptm_mix_kernel(const _Float16* __restrict__ branch,
                     const float*    __restrict__ att,
                     float*          __restrict__ out)
{
    const unsigned total = (unsigned)NB * TT * CH * (HWD / 4);   // 6,422,528
    unsigned q = blockIdx.x * 256u + threadIdx.x;
    if (q >= total) return;
    unsigned hw4 = q % (HWD / 4);
    unsigned tmp = q / (HWD / 4);
    unsigned c   = tmp & 63; tmp >>= 6;
    unsigned t   = tmp & 15;
    unsigned nn  = tmp >> 4;

    float4 o = {0.f, 0.f, 0.f, 0.f};
    #pragma unroll
    for (int b = 0; b < NBR; ++b) {
        float a = att[(nn * 3 + b) * CH + c];
        h4 v = *(const h4*)(branch +
                ((((size_t)b * NB + nn) * CH + c) * TT + t) * HWD + (size_t)hw4 * 4);
        o.x += a * (float)v[0];
        o.y += a * (float)v[1];
        o.z += a * (float)v[2];
        o.w += a * (float)v[3];
    }
    ((float4*)out)[q] = o;
}

// ---------------------------------------------------------------------------
extern "C" void kernel_launch(void* const* d_in, const int* in_sizes, int n_in,
                              void* d_out, int out_size, void* d_ws, size_t ws_size,
                              hipStream_t stream) {
    const float* x      = (const float*)d_in[0];
    const float* conv_w = (const float*)d_in[1];
    const float* conv_b = (const float*)d_in[2];
    const float* fc_w   = (const float*)d_in[3];
    const float* bn_g   = (const float*)d_in[4];
    const float* bn_b   = (const float*)d_in[5];
    const float* bn_m   = (const float*)d_in[6];
    const float* bn_v   = (const float*)d_in[7];
    const float* fcs_w  = (const float*)d_in[8];
    const float* fcs_b  = (const float*)d_in[9];

    // workspace layout (all regions fully overwritten each call):
    //   [0,      6144)   att      : 8*3*64 f32
    //   [8192,   409600) partial  : 196*8*64 f32
    //   [409600, ...)    branch   : 3*8*64*16*3136 f16  (~147 MB)
    float*     att     = (float*)d_ws;
    float*     partial = (float*)((char*)d_ws + 8192);
    _Float16*  branch  = (_Float16*)((char*)d_ws + 409600);

    cptm_conv_kernel<<<NB * NTILE, 128, SMEM_BYTES, stream>>>(
        x, conv_w, conv_b, branch, partial);

    cptm_attn_kernel<<<1, 256, 0, stream>>>(
        partial, fc_w, bn_g, bn_b, bn_m, bn_v, fcs_w, fcs_b, att);

    const unsigned total4 = (unsigned)NB * TT * CH * (HWD / 4);
    cptm_mix_kernel<<<(total4 + 255) / 256, 256, 0, stream>>>(
        branch, att, (float*)d_out);
}
